// GNNEmbeddings_6940667150732
// MI455X (gfx1250) — matrix operand order repaired and checked by reading, not compiled, not run on hardware
//
#include <hip/hip_runtime.h>
#include <hip/hip_bf16.h>

typedef _Float16 v16h __attribute__((ext_vector_type(16)));
typedef _Float16 v8h  __attribute__((ext_vector_type(8)));
typedef _Float16 v2h  __attribute__((ext_vector_type(2)));
typedef float    v8f  __attribute__((ext_vector_type(8)));

#define NNODES 102
#define MPAD   112          // 7 tiles of 16
#define NEDGE  84
#define HD     256          // hidden = embed = 256
#define LH     264          // padded H row stride in halves (264*2=528B, 16B aligned, breaks bank conflicts)
#define SAMPLES_PER_BLOCK 4

static __device__ __forceinline__ v16h cat8(v8h lo, v8h hi) {
    return __builtin_shufflevector(lo, hi, 0,1,2,3,4,5,6,7,8,9,10,11,12,13,14,15);
}

// ---------------- Kernel 1: sparse GCN normalization prep ----------------
// deg[v] = 1 + #incoming(col==v); dinv = 1/sqrt(deg)
// per-edge weight w[e] = dinv[row]*dinv[col]; diagonal dinv2[v] = dinv[v]^2
__global__ void gnn_prep_kernel(const int* __restrict__ ei,  // [2][84]
                                float* __restrict__ w_e,     // [84]
                                float* __restrict__ dinv2)   // [102]
{
    __shared__ int   deg[NNODES];
    __shared__ float dinv[NNODES];
    int t = threadIdx.x;                       // 128 threads
    if (t < NNODES) deg[t] = 1;
    __syncthreads();
    if (t < NEDGE) atomicAdd(&deg[ei[NEDGE + t]], 1);
    __syncthreads();
    if (t < NNODES) dinv[t] = 1.0f / sqrtf((float)deg[t]);
    __syncthreads();
    if (t < NNODES) dinv2[t] = dinv[t] * dinv[t];
    if (t < NEDGE)  w_e[t]   = dinv[ei[t]] * dinv[ei[NEDGE + t]];
}

// ---------------- Kernel 2: W_proj [K=256][N=256] f32 -> Wt [N][K] f16 ----------------
__global__ void gnn_transpose_kernel(const float* __restrict__ W, _Float16* __restrict__ Wt)
{
    __shared__ float tile[16][17];
    int bx = blockIdx.x, by = blockIdx.y;
    int tx = threadIdx.x, ty = threadIdx.y;
    tile[ty][tx] = W[(by*16 + ty)*HD + (bx*16 + tx)];
    __syncthreads();
    Wt[(size_t)(bx*16 + ty)*HD + (by*16 + tx)] = (_Float16)tile[tx][ty];
}

// ---------------- Kernel 3: fused GCN + projection GEMM ----------------
// One block = SAMPLES_PER_BLOCK samples. 256 threads = 8 waves (wave32).
// Wave w owns output N-tiles {w, w+8}; B fragments live in registers across samples.
__global__ void __launch_bounds__(256)
gnn_main_kernel(const float* __restrict__ x,        // [S][204]
                const float* __restrict__ W_gnn,    // [2][256]
                const float* __restrict__ b_gnn,    // [256]
                const float* __restrict__ b_proj,   // [256]
                const int*   __restrict__ ei,       // [2][84]
                const float* __restrict__ w_e,      // [84]
                const float* __restrict__ dinv2,    // [102]
                const _Float16* __restrict__ Wt,    // [256][256] transposed f16
                float* __restrict__ out,            // [S][102][256]
                int total_samples)
{
    __shared__ float Xs[NNODES * 2];
    __shared__ float Ys[NNODES * 2];
    __shared__ __align__(16) _Float16 Hs[MPAD * LH];

    const int t    = threadIdx.x;
    const int wave = t >> 5;
    const int lane = t & 31;
    const int lHi  = lane >> 4;     // half-wave select
    const int lLo  = lane & 15;

    // ---- hoisted per-thread constants for the H-fill phase ----
    // thread owns column pair (cpair, cpair+1) and a 51-row half of H
    const int cpair = (t & 127) * 2;
    const int vbase = (t >> 7) * 51;       // rows 0..50 or 51..101
    const float w0a = W_gnn[cpair],      w0b = W_gnn[cpair + 1];
    const float w1a = W_gnn[HD + cpair], w1b = W_gnn[HD + cpair + 1];
    const float bga = b_gnn[cpair],      bgb = b_gnn[cpair + 1];

    int   er = 0, ec = 0; float we = 0.f;
    if (t < 2 * NEDGE) {
        int e = t >> 1;
        er = ei[e];           // row (source)
        ec = ei[NEDGE + e];   // col (target)
        we = w_e[e];
    }

    // ---- zero padding rows of H once (rows 102..111), packed b32 stores ----
    {
        const int pv = NNODES + (t >> 7) * 5;   // 5 pad rows per half
        #pragma unroll
        for (int i = 0; i < 5; ++i)
            *(v2h*)(Hs + (pv + i) * LH + cpair) = (v2h){(_Float16)0.f, (_Float16)0.f};
    }

    // ---- preload B fragments for this wave's two N-tiles (held in VGPRs) ----
    // B, contiguous k order (same as A): lanes 0-15 -> N=n0+lane, K=k0..k0+15; lanes 16-31 -> K=k0+16..k0+31
    v16h Bf[16];
    float bias[2];
    #pragma unroll
    for (int j = 0; j < 2; ++j) {
        const int col = wave * 16 + j * 128 + lLo;
        const _Float16* wtp = Wt + (size_t)col * HD + lHi * 16;
        #pragma unroll
        for (int k = 0; k < 8; ++k) {
            v8h lo = *(const v8h*)(wtp + k * 32);
            v8h hi = *(const v8h*)(wtp + k * 32 + 8);
            Bf[j * 8 + k] = cat8(lo, hi);
        }
        bias[j] = b_proj[col];
    }

    for (int g = 0; g < SAMPLES_PER_BLOCK; ++g) {
        const size_t s = (size_t)blockIdx.x * SAMPLES_PER_BLOCK + g;
        if (s >= (size_t)total_samples) break;    // uniform per block

        __syncthreads();                          // Hs/Xs/Ys free from previous sample
        // ---- load sample + diagonal term of y = A @ X (pure writes, one phase) ----
        if (t < NNODES * 2) {
            float xv = x[s * (NNODES * 2) + t];
            Xs[t] = xv;
            Ys[t] = dinv2[t >> 1] * xv;
        }
        __syncthreads();
        // ---- sparse edge accumulation (84 weighted edges x 2 channels) ----
        if (t < 2 * NEDGE) {
            int c = t & 1;
            atomicAdd(&Ys[ec * 2 + c], we * Xs[er * 2 + c]); // ds_add_f32
        }
        __syncthreads();

        // ---- H[v][j] = relu(y0*W0[j] + y1*W1[j] + bg[j]) as f16, packed b32 stores ----
        for (int i = 0; i < 51; ++i) {
            const int v = vbase + i;
            const float y0 = Ys[2 * v], y1 = Ys[2 * v + 1];
            const float ha = fmaxf(fmaf(y0, w0a, fmaf(y1, w1a, bga)), 0.f);
            const float hb = fmaxf(fmaf(y0, w0b, fmaf(y1, w1b, bgb)), 0.f);
            *(v2h*)(Hs + v * LH + cpair) = (v2h){(_Float16)ha, (_Float16)hb};
        }
        __syncthreads();

        // ---- GEMM: out[102,256] = H[102,256] @ W_proj[256,256] + b_proj ----
        for (int mt = 0; mt < 7; ++mt) {
            v8f acc0 = {0.f,0.f,0.f,0.f,0.f,0.f,0.f,0.f};
            v8f acc1 = {0.f,0.f,0.f,0.f,0.f,0.f,0.f,0.f};
            // A, contiguous k order (same as B): lanes 0-15 row=m0+lane K:{0-15}; lanes 16-31 K:{16-31}
            const _Float16* hp = Hs + (mt * 16 + lLo) * LH + lHi * 16;
            #pragma unroll
            for (int k = 0; k < 8; ++k) {
                v8h lo = *(const v8h*)(hp + k * 32);
                v8h hi = *(const v8h*)(hp + k * 32 + 8);
                v16h a = cat8(lo, hi);
                acc0 = __builtin_amdgcn_wmma_f32_16x16x32_f16(
                        false, a, false, Bf[k],     (short)0, acc0, false, false);
                acc1 = __builtin_amdgcn_wmma_f32_16x16x32_f16(
                        false, a, false, Bf[8 + k], (short)0, acc1, false, false);
            }
            // C/D layout: VGPR r -> row = m0 + r + 8*lHi, col = n0 + lLo
            const int rbase = mt * 16 + lHi * 8;
            const int col0  = wave * 16 + lLo;
            float* op0 = out + ((size_t)s * NNODES + rbase) * HD + col0;
            #pragma unroll
            for (int r = 0; r < 8; ++r) {
                if (rbase + r < NNODES) {
                    __builtin_nontemporal_store(acc0[r] + bias[0], op0 + (size_t)r * HD);
                    __builtin_nontemporal_store(acc1[r] + bias[1], op0 + (size_t)r * HD + 128);
                }
            }
        }
    }
}

extern "C" void kernel_launch(void* const* d_in, const int* in_sizes, int n_in,
                              void* d_out, int out_size, void* d_ws, size_t ws_size,
                              hipStream_t stream) {
    const float* x      = (const float*)d_in[0];
    const float* W_gnn  = (const float*)d_in[1];
    const float* b_gnn  = (const float*)d_in[2];
    const float* W_proj = (const float*)d_in[3];
    const float* b_proj = (const float*)d_in[4];
    const int*   ei     = (const int*)d_in[5];
    float* out = (float*)d_out;

    // workspace layout: [0..84) w_e f32 | [128..230) dinv2 f32 | byte 1024: Wt f16 [256*256]
    float*     w_e   = (float*)d_ws;
    float*     dinv2 = (float*)d_ws + 128;
    _Float16*  Wt    = (_Float16*)((char*)d_ws + 1024);

    gnn_prep_kernel<<<1, 128, 0, stream>>>(ei, w_e, dinv2);
    gnn_transpose_kernel<<<dim3(16, 16), dim3(16, 16), 0, stream>>>(W_proj, Wt);

    const int S = in_sizes[0] / (NNODES * 2);   // 8192 samples
    const int nblocks = (S + SAMPLES_PER_BLOCK - 1) / SAMPLES_PER_BLOCK;
    gnn_main_kernel<<<nblocks, 256, 0, stream>>>(x, W_gnn, b_gnn, b_proj, ei,
                                                 w_e, dinv2, Wt, out, S);
}
